// FlashRWAttention_33904471834733
// MI455X (gfx1250) — compile-verified, matrix operand order checked by reading, not run on hardware
//
#include <hip/hip_runtime.h>
#include <hip/hip_bf16.h>
#include <math.h>

// ---------------------------------------------------------------------------
// Problem constants (from reference)
// ---------------------------------------------------------------------------
#define NUM_HEADS 64
#define HEAD_SIZE 64
#define HIDDEN    4096
#define NQKV      4224          // HIDDEN + 2*HEAD_SIZE (1 KV head)
#define SEQ       1024
#define BATCH     4
#define TOTAL     4096          // BATCH*SEQ
#define SM_SCALE  0.125f        // 64^-0.5

// ---------------------------------------------------------------------------
// Vector types
// ---------------------------------------------------------------------------
typedef __attribute__((ext_vector_type(16))) __bf16 v16bf;
typedef __attribute__((ext_vector_type(8)))  __bf16 v8bf;
typedef __attribute__((ext_vector_type(4)))  __bf16 bf16x4;
typedef __attribute__((ext_vector_type(8)))  float  v8f;
typedef __attribute__((ext_vector_type(4)))  unsigned int u32x4;
typedef __attribute__((ext_vector_type(4)))  unsigned int v4u;
typedef __attribute__((ext_vector_type(8)))  int v8i;
typedef __attribute__((ext_vector_type(4)))  int v4i;

// Load one 16x32 bf16 A/B fragment from LDS (row-major, ld = 64 bf16).
// Per CDNA5 ISA layout: lanes 0-15 hold K = {0..7, 16..23}, lanes 16-31 hold
// K = {8..15, 24..31} -> two 16-byte ds_load_b128 per lane.
__device__ __forceinline__ v16bf ld_frag(const __bf16* rowp, int ks, int hlf) {
    v8bf a = *(const v8bf*)(rowp + ks + hlf * 8);
    v8bf b = *(const v8bf*)(rowp + ks + 16 + hlf * 8);
    return __builtin_shufflevector(a, b, 0,1,2,3,4,5,6,7,8,9,10,11,12,13,14,15);
}

__device__ __forceinline__ v8f wmma_bf16(v16bf a, v16bf b, v8f c) {
    return __builtin_amdgcn_wmma_f32_16x16x32_bf16(
        /*neg_a=*/false, a, /*neg_b=*/false, b,
        /*c_mod=*/(short)0, c, /*reuse_a=*/false, /*reuse_b=*/false);
}

// ---------------------------------------------------------------------------
// TDM: issue one tensor_load_to_lds moving a 2-D bf16 tile
//   tile_d1 rows x tile_d0 cols, row stride = tensor_d0 elements,
//   packed row-major into LDS at byte offset lds_off.
// D# layout per CDNA5 ISA 8.3/8.4: group0 = {count=1, lds_addr, global_addr,
// type=2}; group1 = {data_size=2B, dims, tile dims, dim0 stride}; groups 2/3
// zero (<=2D tensor). Tracked by TENSORcnt.
// ---------------------------------------------------------------------------
__device__ __forceinline__ void tdm_load_tile_bf16(const __bf16* gbase,
                                                   unsigned lds_off,
                                                   unsigned tensor_d0,
                                                   unsigned tensor_d1,
                                                   unsigned tile_d0,
                                                   unsigned tile_d1) {
    unsigned long long ga = (unsigned long long)(size_t)gbase;
    v4u g0;
    g0[0] = 1u;                                            // count=1 (valid user D#)
    g0[1] = lds_off;                                       // lds_addr (bytes)
    g0[2] = (unsigned)(ga & 0xffffffffu);                  // global_addr[31:0]
    g0[3] = (unsigned)((ga >> 32) & 0x01ffffffu) | (2u << 30); // addr[56:32] | type=2
    v8i g1;
    g1[0] = (int)(1u << 16);                               // data_size=1 (2 bytes)
    g1[1] = (int)((tensor_d0 & 0xffffu) << 16);            // tensor_dim0[15:0]
    g1[2] = (int)((tensor_d0 >> 16) | ((tensor_d1 & 0xffffu) << 16)); // dim0 hi | dim1 lo
    g1[3] = (int)((tensor_d1 >> 16) | (tile_d0 << 16));    // dim1 hi | tile_dim0
    g1[4] = (int)tile_d1;                                  // tile_dim1 (tile_dim2=0)
    g1[5] = (int)tensor_d0;                                // dim0_stride[31:0]
    g1[6] = 0;                                             // dim0_stride hi | dim1_stride lo
    g1[7] = 0;                                             // dim1_stride hi
    v4i z4 = {0, 0, 0, 0};
#if __clang_major__ >= 23
    v8i z8 = {0, 0, 0, 0, 0, 0, 0, 0};
    __builtin_amdgcn_tensor_load_to_lds(g0, g1, z4, z4, z8, 0);
#else
    __builtin_amdgcn_tensor_load_to_lds(g0, g1, z4, z4, 0);
#endif
}

__device__ __forceinline__ unsigned lds_offset_of(const void* p) {
    // low 32 bits of a generic LDS pointer are the offset within the
    // workgroup's LDS allocation (aperture calc, ISA 10.2)
    return (unsigned)(size_t)p;
}

// ---------------------------------------------------------------------------
// fp32 -> bf16 elementwise convert (vectorized x4)
// ---------------------------------------------------------------------------
__global__ void cvt_bf16_kernel(const float* __restrict__ in,
                                __bf16* __restrict__ out, int n4) {
    int i = blockIdx.x * blockDim.x + threadIdx.x;
    if (i < n4) {
        float4 f = *(const float4*)(in + (size_t)i * 4);
        bf16x4 o = { (__bf16)f.x, (__bf16)f.y, (__bf16)f.z, (__bf16)f.w };
        *(bf16x4*)(out + (size_t)i * 4) = o;
    }
}

// ---------------------------------------------------------------------------
// fp32 [R][C] -> bf16 [C][R] tiled transpose-convert (32x32 LDS tile)
// ---------------------------------------------------------------------------
__global__ void transpose_cvt_kernel(const float* __restrict__ in,
                                     __bf16* __restrict__ out, int R, int C) {
    __shared__ float tile[32][33];
    int c0 = blockIdx.x * 32, r0 = blockIdx.y * 32;
    for (int i = threadIdx.y; i < 32; i += 8)
        tile[i][threadIdx.x] = in[(size_t)(r0 + i) * C + c0 + threadIdx.x];
    __syncthreads();
    for (int i = threadIdx.y; i < 32; i += 8)
        out[(size_t)(c0 + i) * R + r0 + threadIdx.x] = (__bf16)tile[threadIdx.x][i];
}

// ---------------------------------------------------------------------------
// bf16 WMMA GEMM:  C[M][N] (f32) = A[M][K] (bf16) * Bt[N][K]^T (bf16)
// 128x128 block, 256 threads = 8 waves (4 row x 2 col), wave tile 32x64,
// K-step 64 (two WMMA k=32 substeps).
// Staging: TDM tensor_load_to_lds, double-buffered. Wave 0 DMAs the A tile,
// wave 1 the B tile for iteration i+1 while all waves run WMMA on buffer i;
// TENSORcnt wait + workgroup barrier publishes the tile.
// ---------------------------------------------------------------------------
__global__ __launch_bounds__(256)
void gemm_bf16_kernel(const __bf16* __restrict__ A,
                      const __bf16* __restrict__ Bt,
                      float* __restrict__ C, int M, int N, int K) {
    __shared__ __align__(16) __bf16 As[2][128 * 64];
    __shared__ __align__(16) __bf16 Bs[2][128 * 64];

    const int tid  = threadIdx.x;
    const int lane = tid & 31, wv = tid >> 5;
    const int hlf  = (lane >> 4) & 1, ln = lane & 15;
    const int wr   = wv & 3;          // wave row (0..3) -> 32 rows
    const int wc   = wv >> 2;         // wave col (0..1) -> 64 cols
    const int m0   = blockIdx.y * 128, n0 = blockIdx.x * 128;

    const v8f zero8 = {0.f,0.f,0.f,0.f,0.f,0.f,0.f,0.f};
    v8f acc[2][4];
    for (int i = 0; i < 2; i++)
        for (int j = 0; j < 4; j++) acc[i][j] = zero8;

    // prologue: DMA first K-tile into buffer 0
    if (wv == 0)
        tdm_load_tile_bf16(A + (size_t)m0 * K, lds_offset_of(&As[0][0]),
                           (unsigned)K, (unsigned)M, 64u, 128u);
    if (wv == 1)
        tdm_load_tile_bf16(Bt + (size_t)n0 * K, lds_offset_of(&Bs[0][0]),
                           (unsigned)K, (unsigned)N, 64u, 128u);
    if (wv < 2) __builtin_amdgcn_s_wait_tensorcnt(0);
    __syncthreads();

    int buf = 0;
    for (int k0 = 0; k0 < K; k0 += 64) {
        const int nb = buf ^ 1;
        if (k0 + 64 < K) {          // DMA next tile while computing this one
            if (wv == 0)
                tdm_load_tile_bf16(A + (size_t)m0 * K + k0 + 64,
                                   lds_offset_of(&As[nb][0]),
                                   (unsigned)K, (unsigned)M, 64u, 128u);
            if (wv == 1)
                tdm_load_tile_bf16(Bt + (size_t)n0 * K + k0 + 64,
                                   lds_offset_of(&Bs[nb][0]),
                                   (unsigned)K, (unsigned)N, 64u, 128u);
        }

        for (int ks = 0; ks < 64; ks += 32) {
            v16bf af[2], bfr[4];
            for (int i = 0; i < 2; i++)
                af[i] = ld_frag(&As[buf][(wr * 32 + i * 16 + ln) * 64], ks, hlf);
            for (int j = 0; j < 4; j++)
                bfr[j] = ld_frag(&Bs[buf][(wc * 64 + j * 16 + ln) * 64], ks, hlf);
            for (int i = 0; i < 2; i++)
                for (int j = 0; j < 4; j++)
                    acc[i][j] = wmma_bf16(af[i], bfr[j], acc[i][j]);
        }

        if (wv < 2) __builtin_amdgcn_s_wait_tensorcnt(0);
        __syncthreads();
        buf = nb;
    }

    // C/D layout: VGPR r -> row r + 8*(lane>=16), col = lane%16
    for (int i = 0; i < 2; i++)
        for (int j = 0; j < 4; j++)
            for (int r = 0; r < 8; r++) {
                int gr = m0 + wr * 32 + i * 16 + r + hlf * 8;
                int gc = n0 + wc * 64 + j * 16 + ln;
                C[(size_t)gr * N + gc] = acc[i][j][r];
            }
}

// ---------------------------------------------------------------------------
// RoPE in-place on f32 qkv buffer: 64 Q heads + 1 K head, rotate (d, d+32)
// ---------------------------------------------------------------------------
__global__ void rope_kernel(float* __restrict__ qkv,
                            const float* __restrict__ cosp,
                            const float* __restrict__ sinp) {
    int idx = blockIdx.x * blockDim.x + threadIdx.x;   // TOTAL * 65 * 32
    if (idx >= TOTAL * 65 * 32) return;
    int t   = idx / (65 * 32);
    int rem = idx % (65 * 32);
    int hh  = rem >> 5, d = rem & 31;
    int col = (hh < 64) ? hh * 64 + d : HIDDEN + d;    // K head lives at col 4096
    float c = cosp[t * 32 + d], s = sinp[t * 32 + d];
    float* p = qkv + (size_t)t * NQKV;
    float x1 = p[col], x2 = p[col + 32];
    p[col]      = x1 * c - x2 * s;
    p[col + 32] = x2 * c + x1 * s;
}

// ---------------------------------------------------------------------------
// Causal flash attention, WMMA bf16, online softmax.
// Grid: (BATCH * SEQ/64, NUM_HEADS). Block: 128 thr = 4 waves; each wave owns
// 16 query rows of a 64-query block, loops over 64-key blocks (causal).
// ---------------------------------------------------------------------------
__global__ __launch_bounds__(128)
void flash_attn_kernel(const float* __restrict__ qkv,
                       __bf16* __restrict__ out) {
    __shared__ __align__(16) __bf16 Qs[64 * 64];       // [s][d]
    __shared__ __align__(16) __bf16 Ks[64 * 64];       // [t][d]
    __shared__ __align__(16) __bf16 Vt[64 * 64];       // [d][t]  (transposed)
    __shared__ __align__(16) __bf16 Ps[4][16 * 64];    // per-wave P scratch

    const int tid = threadIdx.x, lane = tid & 31, wv = tid >> 5;
    const int hlf = (lane >> 4) & 1, ln = lane & 15;
    const int qb = blockIdx.x & 15, batch = blockIdx.x >> 4, h = blockIdx.y;
    const int tok0 = batch * SEQ + qb * 64;

    // ---- stage Q block (f32 -> bf16) ----
    for (int c = tid; c < 1024; c += 128) {
        int row = c >> 4, f4 = (c & 15) << 2;
        float4 v = *(const float4*)(qkv + (size_t)(tok0 + row) * NQKV + h * 64 + f4);
        bf16x4 o = { (__bf16)v.x, (__bf16)v.y, (__bf16)v.z, (__bf16)v.w };
        *(bf16x4*)(Qs + row * 64 + f4) = o;
    }
    __syncthreads();
    v16bf qa0 = ld_frag(Qs + (wv * 16 + ln) * 64, 0, hlf);
    v16bf qa1 = ld_frag(Qs + (wv * 16 + ln) * 64, 32, hlf);

    const v8f zero8 = {0.f,0.f,0.f,0.f,0.f,0.f,0.f,0.f};
    float mrun[8], lrun[8];
    v8f accO[4];
    for (int r = 0; r < 8; r++) { mrun[r] = -1e30f; lrun[r] = 0.f; }
    for (int j = 0; j < 4; j++) accO[j] = zero8;

    for (int kb = 0; kb <= qb; kb++) {                 // causal: skip upper blocks
        __syncthreads();
        int ktok0 = batch * SEQ + kb * 64;
        // ---- stage K (row-major) and V (transposed) ----
        for (int c = tid; c < 1024; c += 128) {
            int row = c >> 4, f4 = (c & 15) << 2;
            const float* base = qkv + (size_t)(ktok0 + row) * NQKV + HIDDEN;
            float4 kv4 = *(const float4*)(base + f4);
            bf16x4 ko = { (__bf16)kv4.x, (__bf16)kv4.y, (__bf16)kv4.z, (__bf16)kv4.w };
            *(bf16x4*)(Ks + row * 64 + f4) = ko;
            float4 vv4 = *(const float4*)(base + 64 + f4);
            Vt[(f4 + 0) * 64 + row] = (__bf16)vv4.x;
            Vt[(f4 + 1) * 64 + row] = (__bf16)vv4.y;
            Vt[(f4 + 2) * 64 + row] = (__bf16)vv4.z;
            Vt[(f4 + 3) * 64 + row] = (__bf16)vv4.w;
        }
        __syncthreads();

        // ---- S = Q K^T * scale (4 tiles of 16x16 per wave) ----
        v8f sc[4];
        for (int j = 0; j < 4; j++) {
            v16bf kf0 = ld_frag(Ks + (j * 16 + ln) * 64, 0, hlf);
            v16bf kf1 = ld_frag(Ks + (j * 16 + ln) * 64, 32, hlf);
            v8f s = zero8;
            s = wmma_bf16(qa0, kf0, s);
            s = wmma_bf16(qa1, kf1, s);
            s = s * SM_SCALE;
            if (kb == qb) {                            // diagonal block mask
                for (int r = 0; r < 8; r++) {
                    int srow = wv * 16 + r + hlf * 8;
                    int tcol = j * 16 + ln;
                    if (tcol > srow) s[r] = -1e30f;
                }
            }
            sc[j] = s;
        }

        // ---- online softmax: row max via half-wave shfl_xor ----
        float corr[8];
        for (int r = 0; r < 8; r++) {
            float v = fmaxf(fmaxf(sc[0][r], sc[1][r]), fmaxf(sc[2][r], sc[3][r]));
            for (int msk = 8; msk >= 1; msk >>= 1)
                v = fmaxf(v, __shfl_xor(v, msk, 32));
            float mnew = fmaxf(mrun[r], v);
            corr[r] = __expf(mrun[r] - mnew);
            mrun[r] = mnew;
            lrun[r] *= corr[r];
        }
        for (int j = 0; j < 4; j++)
            for (int r = 0; r < 8; r++) accO[j][r] *= corr[r];

        // ---- P = exp(S - m); write bf16 P to wave-private LDS (A-layout src) ----
        for (int j = 0; j < 4; j++)
            for (int r = 0; r < 8; r++) {
                float p = __expf(sc[j][r] - mrun[r]);
                sc[j][r] = p;
                Ps[wv][(r + hlf * 8) * 64 + j * 16 + ln] = (__bf16)p;
            }
        for (int r = 0; r < 8; r++) {
            float s = sc[0][r] + sc[1][r] + sc[2][r] + sc[3][r];
            for (int msk = 8; msk >= 1; msk >>= 1)
                s += __shfl_xor(s, msk, 32);
            lrun[r] += s;
        }

        // ---- O += P V (contract over t=64, two k=32 substeps) ----
        v16bf pa0 = ld_frag(Ps[wv] + ln * 64, 0, hlf);
        v16bf pa1 = ld_frag(Ps[wv] + ln * 64, 32, hlf);
        for (int jd = 0; jd < 4; jd++) {
            v16bf vf0 = ld_frag(Vt + (jd * 16 + ln) * 64, 0, hlf);
            v16bf vf1 = ld_frag(Vt + (jd * 16 + ln) * 64, 32, hlf);
            accO[jd] = wmma_bf16(pa0, vf0, accO[jd]);
            accO[jd] = wmma_bf16(pa1, vf1, accO[jd]);
        }
    }

    // ---- epilogue: O / l -> bf16 [T][HIDDEN] feeding dense GEMM ----
    for (int jd = 0; jd < 4; jd++)
        for (int r = 0; r < 8; r++) {
            int m   = r + hlf * 8;
            int tok = tok0 + wv * 16 + m;
            int col = h * 64 + jd * 16 + ln;
            out[(size_t)tok * HIDDEN + col] = (__bf16)(accO[jd][r] / lrun[r]);
        }
}

// ---------------------------------------------------------------------------
// Host launcher
// ---------------------------------------------------------------------------
extern "C" void kernel_launch(void* const* d_in, const int* in_sizes, int n_in,
                              void* d_out, int out_size, void* d_ws, size_t ws_size,
                              hipStream_t stream) {
    const float* hidden = (const float*)d_in[0];   // [4096][4096]
    const float* cosp   = (const float*)d_in[1];   // [4096][32]
    const float* sinp   = (const float*)d_in[2];   // [4096][32]
    const float* wqkv   = (const float*)d_in[3];   // [4096][4224]
    const float* wdense = (const float*)d_in[4];   // [4096][4096]
    // d_in[5] = cu_seqlen (regular batching, unused)

    char* ws = (char*)d_ws;
    __bf16* hidB   = (__bf16*)ws;  ws += (size_t)TOTAL * HIDDEN * 2;
    __bf16* wqkvT  = (__bf16*)ws;  ws += (size_t)NQKV * HIDDEN * 2;   // [N][K]
    __bf16* wdenT  = (__bf16*)ws;  ws += (size_t)HIDDEN * HIDDEN * 2; // [N][K]
    float*  qkvBuf = (float*)ws;   ws += (size_t)TOTAL * NQKV * 4;
    __bf16* attnB  = (__bf16*)ws;  // [TOTAL][HIDDEN] bf16

    // convert activations + pre-transpose weights to bf16 [N][K]
    cvt_bf16_kernel<<<(TOTAL * HIDDEN / 4 + 255) / 256, 256, 0, stream>>>(
        hidden, hidB, TOTAL * HIDDEN / 4);
    transpose_cvt_kernel<<<dim3(NQKV / 32, HIDDEN / 32), dim3(32, 8), 0, stream>>>(
        wqkv, wqkvT, HIDDEN, NQKV);
    transpose_cvt_kernel<<<dim3(HIDDEN / 32, HIDDEN / 32), dim3(32, 8), 0, stream>>>(
        wdense, wdenT, HIDDEN, HIDDEN);

    // QKV projection (f32 accumulate)
    gemm_bf16_kernel<<<dim3(NQKV / 128, TOTAL / 128), 256, 0, stream>>>(
        hidB, wqkvT, qkvBuf, TOTAL, NQKV, HIDDEN);

    // RoPE on Q (64 heads) + K (1 head), in-place
    rope_kernel<<<(TOTAL * 65 * 32 + 255) / 256, 256, 0, stream>>>(
        qkvBuf, cosp, sinp);

    // causal flash attention -> bf16 activations
    flash_attn_kernel<<<dim3(BATCH * (SEQ / 64), NUM_HEADS), 128, 0, stream>>>(
        qkvBuf, attnB);

    // dense output projection -> f32 d_out
    gemm_bf16_kernel<<<dim3(HIDDEN / 128, TOTAL / 128), 256, 0, stream>>>(
        attnB, wdenT, (float*)d_out, TOTAL, HIDDEN, HIDDEN);
}
